// StackSAModuleMSG_85761906966880
// MI455X (gfx1250) — matrix operand order repaired and checked by reading, not compiled, not run on hardware
//
#include <hip/hip_runtime.h>
#include <hip/hip_bf16.h>

typedef __attribute__((ext_vector_type(16))) _Float16 v16h;
typedef __attribute__((ext_vector_type(8)))  _Float16 v8h;
typedef __attribute__((ext_vector_type(8)))  float    v8f;

union F16x16 { v16h v; v8h h[2]; };

#define Bsz 4
#define NB 8192
#define MB 2048
#define NQ (Bsz * MB)          // 8192 queries
#define CIN 64
#define NS0 16
#define NS1 32
#define R0SQ 0.04f
#define R1SQ 0.16f
#define EPSBN 1e-5f

#define NROWS0 (NQ * NS0)      // 131072
#define NROWS1 (NQ * NS1)      // 262144

// ---- workspace layout (bytes) ----
#define OFF_IDX0  0u
#define OFF_IDX1  524288u
#define OFF_W0H0  1572864u     // 128*96 f16
#define OFF_W0H1  1597440u
#define OFF_W1H0  1622016u     // 128*128 f16
#define OFF_W1H1  1654784u
#define OFF_SUMS  1687552u     // 4 pairs x (sum[128], sq[128]) f32
#define OFF_AB    1691648u     // 4 pairs x (a[128], b[128]) f32
#define OFF_X0    1695744u     // 262144*128 f16
#define OFF_X1    68804608u    // 262144*128 f16

// =============================== prep kernels ===============================

__global__ void zero_stats_kernel(float* sums) {
  int i = threadIdx.x;
  for (int k = i; k < 1024; k += 256) sums[k] = 0.f;
}

__global__ void copy_newxyz_kernel(const float* __restrict__ nx, float* __restrict__ out) {
  int i = blockIdx.x * blockDim.x + threadIdx.x;
  if (i < NQ * 3) out[i] = nx[i];
}

__global__ void convw_kernel(const float* __restrict__ w00, const float* __restrict__ w01,
                             const float* __restrict__ w10, const float* __restrict__ w11,
                             _Float16* __restrict__ w0h0, _Float16* __restrict__ w0h1,
                             _Float16* __restrict__ w1h0, _Float16* __restrict__ w1h1) {
  int i = blockIdx.x * blockDim.x + threadIdx.x;
  if (i < 12288) {                       // scale0 layer0: 128 x 96 (pad 67->96)
    int o = i / 96, k = i % 96;
    w0h0[i] = (_Float16)(k < 67 ? w00[o * 67 + k] : 0.f);
  } else if (i < 24576) {                // scale1 layer0
    int j = i - 12288; int o = j / 96, k = j % 96;
    w0h1[j] = (_Float16)(k < 67 ? w10[o * 67 + k] : 0.f);
  } else if (i < 40960) {                // scale0 layer1: 128 x 128
    int j = i - 24576;
    w1h0[j] = (_Float16)w01[j];
  } else if (i < 57344) {                // scale1 layer1
    int j = i - 40960;
    w1h1[j] = (_Float16)w11[j];
  }
}

// ====================== fused dual-radius ball query ========================
// 1 thread per query; block of 256 queries stays inside one batch.
// Points streamed through LDS in 1024-point tiles; one pass fills both scales.
__global__ __launch_bounds__(256)
void ballquery_kernel(const float* __restrict__ xyz, const float* __restrict__ new_xyz,
                      int* __restrict__ idx0, int* __restrict__ idx1) {
  __shared__ float s[1024 * 3];
  int tid = threadIdx.x;
  int m = blockIdx.x * 256 + tid;              // query index (0..8191)
  int batch = m >> 11;                          // 2048 queries per batch
  float qx = new_xyz[m * 3 + 0];
  float qy = new_xyz[m * 3 + 1];
  float qz = new_xyz[m * 3 + 2];
  const float* pbase = xyz + (size_t)batch * NB * 3;

  int cnt0 = 0, cnt1 = 0, first0 = -1, first1 = -1;
  for (int tile = 0; tile < 8; ++tile) {
    __syncthreads();
    #pragma unroll
    for (int k = 0; k < 12; ++k) {
      int off = k * 256 + tid;
      s[off] = pbase[tile * 3072 + off];
    }
    __syncthreads();
    if (cnt1 == NS1 && cnt0 == NS0) continue;   // done, still join barriers
    for (int j = 0; j < 1024; ++j) {
      float dx = s[j * 3 + 0] - qx;
      float dy = s[j * 3 + 1] - qy;
      float dz = s[j * 3 + 2] - qz;
      float d2 = dx * dx + dy * dy + dz * dz;
      if (d2 < R1SQ) {
        int pi = tile * 1024 + j;               // batch-local point index
        if (cnt1 < NS1) { idx1[m * NS1 + cnt1] = pi; if (cnt1 == 0) first1 = pi; ++cnt1; }
        if (d2 < R0SQ && cnt0 < NS0) { idx0[m * NS0 + cnt0] = pi; if (cnt0 == 0) first0 = pi; ++cnt0; }
        if (cnt1 == NS1 && cnt0 == NS0) break;
      }
    }
  }
  for (int k = cnt0; k < NS0; ++k) idx0[m * NS0 + k] = first0;  // -1 == empty -> zero rows
  for (int k = cnt1; k < NS1; ++k) idx1[m * NS1 + k] = first1;
}

// ============================ WMMA GEMM layer 0 =============================
// A: gathered (rel_xyz | feat) rows, K padded to 96. B: W0 f16 row-major per
// out-channel. C: rows x 128 -> f16 Xout + per-channel sum/sumsq for BN.
#define KP0 96
#define AS0 104   // LDS row stride (halves), conflict padding

__global__ __launch_bounds__(256)
void gemm0_kernel(const float* __restrict__ xyz, const float* __restrict__ feat,
                  const float* __restrict__ new_xyz, const int* __restrict__ idxbuf,
                  const _Float16* __restrict__ wh, _Float16* __restrict__ Xout,
                  float* __restrict__ gsum, float* __restrict__ gsq,
                  int ns_shift, int nrows) {
  __shared__ _Float16 As[128 * AS0];
  __shared__ _Float16 Bs[128 * KP0];
  __shared__ float ssum[128], ssq[128];

  int tid = threadIdx.x;
  int wave = tid >> 5, lane = tid & 31;
  int rowbase = blockIdx.x * 128;

  if (tid < 128) { ssum[tid] = 0.f; ssq[tid] = 0.f; }
  for (int i = tid; i < 128 * KP0; i += 256) Bs[i] = wh[i];

  // gather A rows (67 dims, pad to 96), f32 -> f16 into LDS
  {
    int r_local = tid >> 1;
    int cbeg = (tid & 1) * 48;
    int r = rowbase + r_local;
    int id = (r < nrows) ? idxbuf[r] : -1;
    if (id >= 0) {
      int m = r >> ns_shift;
      int gid = ((m >> 11) << 13) + id;         // batch*NB + local idx
      const float* fp = feat + (size_t)gid * CIN;
      for (int c = cbeg; c < cbeg + 48; ++c) {
        float v;
        if (c < 3)       v = xyz[gid * 3 + c] - new_xyz[m * 3 + c];
        else if (c < 67) v = fp[c - 3];
        else             v = 0.f;
        As[r_local * AS0 + c] = (_Float16)v;
      }
    } else {
      for (int c = cbeg; c < cbeg + 48; ++c) As[r_local * AS0 + c] = (_Float16)0.f;
    }
  }
  __syncthreads();

  v8f acc[8];
  #pragma unroll
  for (int t = 0; t < 8; ++t)
    #pragma unroll
    for (int v = 0; v < 8; ++v) acc[t][v] = 0.f;

  int arow = wave * 16 + (lane & 15);
  int a_kb = (lane < 16) ? 0 : 8;               // A: K lo/hi per ISA layout
  int b_col = lane & 15;
  int b_kb = (lane < 16) ? 0 : 8;               // B: K lo/hi per ISA layout

  #pragma unroll
  for (int ks = 0; ks < KP0; ks += 32) {
    F16x16 a;
    const _Float16* ap = &As[arow * AS0 + ks + a_kb];
    a.h[0] = *(const v8h*)ap;
    a.h[1] = *(const v8h*)(ap + 16);
    #pragma unroll
    for (int t = 0; t < 8; ++t) {
      F16x16 b;
      const _Float16* bp = &Bs[(t * 16 + b_col) * KP0 + ks + b_kb];
      b.h[0] = *(const v8h*)bp;
      b.h[1] = *(const v8h*)(bp + 16);
      acc[t] = __builtin_amdgcn_wmma_f32_16x16x32_f16(
          false, a.v, false, b.v, (short)0, acc[t], false, false);
    }
  }

  // store f16 activations + block-local BN partial sums
  int mbase = (lane < 16) ? 0 : 8;
  #pragma unroll
  for (int t = 0; t < 8; ++t) {
    int ch = t * 16 + (lane & 15);
    float s = 0.f, q = 0.f;
    #pragma unroll
    for (int v = 0; v < 8; ++v) {
      float val = acc[t][v];
      int r = rowbase + wave * 16 + mbase + v;
      if (r < nrows) Xout[(size_t)r * 128 + ch] = (_Float16)val;
      s += val; q += val * val;
    }
    atomicAdd(&ssum[ch], s);
    atomicAdd(&ssq[ch], q);
  }
  __syncthreads();
  if (tid < 128) {
    atomicAdd(&gsum[tid], ssum[tid]);
    atomicAdd(&gsq[tid], ssq[tid]);
  }
}

// ===================== BN -> stats fold =====================
__global__ void stats_kernel(const float* __restrict__ gsum, const float* __restrict__ gsq,
                             const float* __restrict__ gamma, const float* __restrict__ beta,
                             float* __restrict__ a_out, float* __restrict__ b_out, float invN) {
  int c = threadIdx.x;                          // 128 threads
  float mean = gsum[c] * invN;
  float var  = gsq[c] * invN - mean * mean;
  float a = gamma[c] * rsqrtf(var + EPSBN);
  a_out[c] = a;
  b_out[c] = beta[c] - mean * a;
}

// ============================ WMMA GEMM layer 1 =============================
// A rows = relu(a*X0 + b) from f16, K = 128.
#define KP1 128
#define AS1 136

__global__ __launch_bounds__(256)
void gemm1_kernel(const _Float16* __restrict__ Xin, const float* __restrict__ ab_a,
                  const float* __restrict__ ab_b, const _Float16* __restrict__ wh,
                  _Float16* __restrict__ Xout, float* __restrict__ gsum,
                  float* __restrict__ gsq, int nrows) {
  __shared__ _Float16 As[128 * AS1];
  __shared__ _Float16 Bs[128 * KP1];
  __shared__ float ssum[128], ssq[128], sa[128], sb[128];

  int tid = threadIdx.x;
  int wave = tid >> 5, lane = tid & 31;
  int rowbase = blockIdx.x * 128;

  if (tid < 128) { sa[tid] = ab_a[tid]; sb[tid] = ab_b[tid]; ssum[tid] = 0.f; ssq[tid] = 0.f; }
  __syncthreads();

  for (int i = tid; i < 128 * KP1; i += 256) Bs[i] = wh[i];

  {
    int r_local = tid >> 1;
    int cbeg = (tid & 1) * 64;
    int r = rowbase + r_local;
    if (r < nrows) {
      const _Float16* xp = Xin + (size_t)r * 128;
      for (int c = cbeg; c < cbeg + 64; ++c) {
        float v = sa[c] * (float)xp[c] + sb[c];
        As[r_local * AS1 + c] = (_Float16)fmaxf(v, 0.f);
      }
    } else {
      for (int c = cbeg; c < cbeg + 64; ++c) As[r_local * AS1 + c] = (_Float16)0.f;
    }
  }
  __syncthreads();

  v8f acc[8];
  #pragma unroll
  for (int t = 0; t < 8; ++t)
    #pragma unroll
    for (int v = 0; v < 8; ++v) acc[t][v] = 0.f;

  int arow = wave * 16 + (lane & 15);
  int a_kb = (lane < 16) ? 0 : 8;
  int b_col = lane & 15;
  int b_kb = (lane < 16) ? 0 : 8;

  #pragma unroll
  for (int ks = 0; ks < KP1; ks += 32) {
    F16x16 a;
    const _Float16* ap = &As[arow * AS1 + ks + a_kb];
    a.h[0] = *(const v8h*)ap;
    a.h[1] = *(const v8h*)(ap + 16);
    #pragma unroll
    for (int t = 0; t < 8; ++t) {
      F16x16 b;
      const _Float16* bp = &Bs[(t * 16 + b_col) * KP1 + ks + b_kb];
      b.h[0] = *(const v8h*)bp;
      b.h[1] = *(const v8h*)(bp + 16);
      acc[t] = __builtin_amdgcn_wmma_f32_16x16x32_f16(
          false, a.v, false, b.v, (short)0, acc[t], false, false);
    }
  }

  int mbase = (lane < 16) ? 0 : 8;
  #pragma unroll
  for (int t = 0; t < 8; ++t) {
    int ch = t * 16 + (lane & 15);
    float s = 0.f, q = 0.f;
    #pragma unroll
    for (int v = 0; v < 8; ++v) {
      float val = acc[t][v];
      int r = rowbase + wave * 16 + mbase + v;
      if (r < nrows) Xout[(size_t)r * 128 + ch] = (_Float16)val;
      s += val; q += val * val;
    }
    atomicAdd(&ssum[ch], s);
    atomicAdd(&ssq[ch], q);
  }
  __syncthreads();
  if (tid < 128) {
    atomicAdd(&gsum[tid], ssum[tid]);
    atomicAdd(&gsq[tid], ssq[tid]);
  }
}

// ================= BN + ReLU + max over samples -> output ==================
__global__ __launch_bounds__(128)
void maxpool_kernel(const _Float16* __restrict__ Xin, const float* __restrict__ ab_a,
                    const float* __restrict__ ab_b, float* __restrict__ out,
                    int ns_shift, int chan_off) {
  int m = blockIdx.x;                            // query
  int c = threadIdx.x;                           // channel
  int ns = 1 << ns_shift;
  float a = ab_a[c], b = ab_b[c];
  const _Float16* xp = Xin + (((size_t)m << ns_shift) * 128) + c;
  float mx = -3.4e38f;
  for (int n = 0; n < ns; ++n)
    mx = fmaxf(mx, a * (float)xp[(size_t)n * 128] + b);
  mx = fmaxf(mx, 0.f);                           // relu commutes with max
  out[NQ * 3 + (size_t)m * 256 + chan_off + c] = mx;
}

// ================================ launch ===================================
extern "C" void kernel_launch(void* const* d_in, const int* in_sizes, int n_in,
                              void* d_out, int out_size, void* d_ws, size_t ws_size,
                              hipStream_t stream) {
  const float* xyz      = (const float*)d_in[0];
  const float* feat     = (const float*)d_in[1];
  const float* new_xyz  = (const float*)d_in[2];
  const float* w00 = (const float*)d_in[5];
  const float* g00 = (const float*)d_in[6];
  const float* b00 = (const float*)d_in[7];
  const float* w01 = (const float*)d_in[8];
  const float* g01 = (const float*)d_in[9];
  const float* b01 = (const float*)d_in[10];
  const float* w10 = (const float*)d_in[11];
  const float* g10 = (const float*)d_in[12];
  const float* b10 = (const float*)d_in[13];
  const float* w11 = (const float*)d_in[14];
  const float* g11 = (const float*)d_in[15];
  const float* b11 = (const float*)d_in[16];
  float* out = (float*)d_out;
  char* ws = (char*)d_ws;

  int*       idx0 = (int*)(ws + OFF_IDX0);
  int*       idx1 = (int*)(ws + OFF_IDX1);
  _Float16*  w0h0 = (_Float16*)(ws + OFF_W0H0);
  _Float16*  w0h1 = (_Float16*)(ws + OFF_W0H1);
  _Float16*  w1h0 = (_Float16*)(ws + OFF_W1H0);
  _Float16*  w1h1 = (_Float16*)(ws + OFF_W1H1);
  float*     sums = (float*)(ws + OFF_SUMS);    // 4 pairs x 256 floats
  float*     ab   = (float*)(ws + OFF_AB);      // 4 pairs x 256 floats
  _Float16*  X0   = (_Float16*)(ws + OFF_X0);
  _Float16*  X1   = (_Float16*)(ws + OFF_X1);

  auto SUM = [&](int p) { return sums + p * 256; };
  auto SQ  = [&](int p) { return sums + p * 256 + 128; };
  auto AA  = [&](int p) { return ab + p * 256; };
  auto BB  = [&](int p) { return ab + p * 256 + 128; };

  zero_stats_kernel<<<1, 256, 0, stream>>>(sums);
  copy_newxyz_kernel<<<(NQ * 3 + 255) / 256, 256, 0, stream>>>(new_xyz, out);
  convw_kernel<<<(57344 + 255) / 256, 256, 0, stream>>>(w00, w01, w10, w11, w0h0, w0h1, w1h0, w1h1);
  ballquery_kernel<<<NQ / 256, 256, 0, stream>>>(xyz, new_xyz, idx0, idx1);

  // ---- scale 0 (r=0.2, ns=16) ----
  gemm0_kernel<<<NROWS0 / 128, 256, 0, stream>>>(xyz, feat, new_xyz, idx0, w0h0,
                                                 X0, SUM(0), SQ(0), 4, NROWS0);
  stats_kernel<<<1, 128, 0, stream>>>(SUM(0), SQ(0), g00, b00, AA(0), BB(0), 1.f / NROWS0);
  gemm1_kernel<<<NROWS0 / 128, 256, 0, stream>>>(X0, AA(0), BB(0), w1h0,
                                                 X1, SUM(1), SQ(1), NROWS0);
  stats_kernel<<<1, 128, 0, stream>>>(SUM(1), SQ(1), g01, b01, AA(1), BB(1), 1.f / NROWS0);
  maxpool_kernel<<<NQ, 128, 0, stream>>>(X1, AA(1), BB(1), out, 4, 0);

  // ---- scale 1 (r=0.4, ns=32) ----
  gemm0_kernel<<<NROWS1 / 128, 256, 0, stream>>>(xyz, feat, new_xyz, idx1, w0h1,
                                                 X0, SUM(2), SQ(2), 5, NROWS1);
  stats_kernel<<<1, 128, 0, stream>>>(SUM(2), SQ(2), g10, b10, AA(2), BB(2), 1.f / NROWS1);
  gemm1_kernel<<<NROWS1 / 128, 256, 0, stream>>>(X0, AA(2), BB(2), w1h1,
                                                 X1, SUM(3), SQ(3), NROWS1);
  stats_kernel<<<1, 128, 0, stream>>>(SUM(3), SQ(3), g11, b11, AA(3), BB(3), 1.f / NROWS1);
  maxpool_kernel<<<NQ, 128, 0, stream>>>(X1, AA(3), BB(3), out, 5, 128);
}
